// Attention_74955769250234
// MI455X (gfx1250) — matrix operand order repaired and checked by reading, not compiled, not run on hardware
//
#include <hip/hip_runtime.h>
#include <hip/hip_bf16.h>

typedef __attribute__((ext_vector_type(16))) _Float16 v16h;
typedef __attribute__((ext_vector_type(8)))  _Float16 v8h;
typedef __attribute__((ext_vector_type(8)))  float    v8f;

#define CHN    256
#define NHD    4
#define DHV    64
#define DKQ    32
#define QKVC   512
#define HW     1600
#define NB     4
#define BN_EPS 1e-3f

// ---------------------------------------------------------------------------
// 1) Fold BN into weights/biases; convert GEMM weights to f16.
// ---------------------------------------------------------------------------
__global__ void prep_kernel(const float* w_qkv, const float* g_qkv, const float* b_qkv,
                            const float* m_qkv, const float* v_qkv,
                            const float* w_dw,  const float* g_dw,  const float* b_dw,
                            const float* m_dw,  const float* v_dw,
                            const float* w_proj,const float* g_proj,const float* b_proj,
                            const float* m_proj,const float* v_proj,
                            _Float16* wqkv_h, float* bias_qkv,
                            _Float16* wproj_h, float* bias_proj,
                            float* wdwf, float* bias_dw) {
  int t = blockIdx.x * blockDim.x + threadIdx.x;
  if (t < QKVC * CHN) {
    int co = t / CHN;
    float s = g_qkv[co] * rsqrtf(v_qkv[co] + BN_EPS);
    wqkv_h[t] = (_Float16)(w_qkv[t] * s);
    return;
  }
  t -= QKVC * CHN;
  if (t < CHN * CHN) {
    int co = t / CHN;
    float s = g_proj[co] * rsqrtf(v_proj[co] + BN_EPS);
    wproj_h[t] = (_Float16)(w_proj[t] * s);
    return;
  }
  t -= CHN * CHN;
  if (t < CHN * 9) {
    int c = t / 9;
    wdwf[t] = w_dw[t] * (g_dw[c] * rsqrtf(v_dw[c] + BN_EPS));
    return;
  }
  t -= CHN * 9;
  if (t < QKVC) {
    float s = g_qkv[t] * rsqrtf(v_qkv[t] + BN_EPS);
    bias_qkv[t] = b_qkv[t] - m_qkv[t] * s;
    return;
  }
  t -= QKVC;
  if (t < CHN) {
    float s = g_proj[t] * rsqrtf(v_proj[t] + BN_EPS);
    bias_proj[t] = b_proj[t] - m_proj[t] * s;
    return;
  }
  t -= CHN;
  if (t < CHN) {
    float s = g_dw[t] * rsqrtf(v_dw[t] + BN_EPS);
    bias_dw[t] = b_dw[t] - m_dw[t] * s;
    return;
  }
}

// ---------------------------------------------------------------------------
// 2) Pack x into WMMA B-operand lane layout (f16).
//    index = (((b*8 + kt)*100 + nt)*32 + lane)*16 + j
//    element: K(ci) = kt*32 + (j>=8)*16 + (lane>=16)*8 + (j&7) ; N(n) = nt*16 + (lane&15)
// ---------------------------------------------------------------------------
__global__ void pack_x_kernel(const float* x, _Float16* xB) {
  int t = blockIdx.x * blockDim.x + threadIdx.x;
  int j = t & 15, lane = (t >> 4) & 31, rest = t >> 9;
  int nt = rest % 100; rest /= 100;
  int kt = rest % 8;  int b = rest / 8;
  int ci = kt * 32 + ((j < 8) ? 0 : 16) + ((lane >> 4) << 3) + (j & 7);
  int n  = nt * 16 + (lane & 15);
  xB[t] = (_Float16)x[(b * CHN + ci) * HW + n];
}

// ---------------------------------------------------------------------------
// 3) QKV GEMM: (512x256) x (256x1600) per batch via WMMA f16->f32.
//    One wave per 16x16 D tile, K in 8 chunks of 32.
// ---------------------------------------------------------------------------
__global__ void __launch_bounds__(128) qkv_gemm_kernel(const _Float16* __restrict__ wh,
                                                       const _Float16* __restrict__ xB,
                                                       const float* __restrict__ bias,
                                                       float* __restrict__ qkv) {
  int lane = threadIdx.x & 31;
  int wid  = blockIdx.x * 4 + (threadIdx.x >> 5);
  int nt   = wid % 100;
  int cot  = (wid / 100) % 32;
  int b    = wid / 3200;
  int r0 = lane & 15, half = lane >> 4;
  v8f c = {};
#pragma unroll
  for (int kt = 0; kt < 8; ++kt) {
    const _Float16* ap = wh + (cot * 16 + r0) * CHN + kt * 32 + half * 8;
    v8h a0 = *(const v8h*)(ap);
    v8h a1 = *(const v8h*)(ap + 16);
    v16h a;
#pragma unroll
    for (int i = 0; i < 8; ++i) { a[i] = a0[i]; a[i + 8] = a1[i]; }
    v16h bm = *(const v16h*)(xB + (((b * 8 + kt) * 100) + nt) * 512 + lane * 16);
    c = __builtin_amdgcn_wmma_f32_16x16x32_f16(false, a, false, bm, (short)0, c, false, false);
  }
#pragma unroll
  for (int r = 0; r < 8; ++r) {
    int co = cot * 16 + r + half * 8;
    qkv[(b * QKVC + co) * HW + nt * 16 + r0] = c[r] + bias[co];
  }
}

// ---------------------------------------------------------------------------
// 4) Repack q (A layout, scale folded), k (B layout), v (B layout).
// ---------------------------------------------------------------------------
__global__ void repack_qkv_kernel(const float* __restrict__ qkv,
                                  _Float16* qA, _Float16* kB, _Float16* vB) {
  int t = blockIdx.x * blockDim.x + threadIdx.x;
  const float scale = 0.17677669529663687f;  // 32^-0.5
  if (t < 16 * 100 * 512) {          // qA
    int j = t & 15, lane = (t >> 4) & 31, rest = t >> 9;
    int nt = rest % 100, bh = rest / 100;
    int d = ((j < 8) ? 0 : 16) + ((lane >> 4) << 3) + (j & 7);
    int n = nt * 16 + (lane & 15);
    int b = bh >> 2, h = bh & 3;
    qA[t] = (_Float16)(qkv[(b * QKVC + h * 128 + d) * HW + n] * scale);
    return;
  }
  t -= 16 * 100 * 512;
  if (t < 16 * 100 * 512) {          // kB
    int j = t & 15, lane = (t >> 4) & 31, rest = t >> 9;
    int mt = rest % 100, bh = rest / 100;
    int d = ((j < 8) ? 0 : 16) + ((lane >> 4) << 3) + (j & 7);
    int m = mt * 16 + (lane & 15);
    int b = bh >> 2, h = bh & 3;
    kB[t] = (_Float16)qkv[(b * QKVC + h * 128 + 32 + d) * HW + m];
    return;
  }
  t -= 16 * 100 * 512;
  if (t < 16 * 50 * 4 * 512) {       // vB
    int j = t & 15, lane = (t >> 4) & 31, rest = t >> 9;
    int dt = rest % 4; rest /= 4;
    int mc = rest % 50; int bh = rest / 50;
    int d = dt * 16 + (lane & 15);
    int m = mc * 32 + ((j < 8) ? 0 : 16) + ((lane >> 4) << 3) + (j & 7);
    int b = bh >> 2, h = bh & 3;
    vB[t] = (_Float16)qkv[(b * QKVC + h * 128 + 64 + d) * HW + m];
  }
}

// ---------------------------------------------------------------------------
// 5) Flash attention: one wave per 16-query tile.
//    Per 32 keys: 2x QK^T WMMA, online softmax (half-wave shuffles),
//    P -> LDS -> A-layout, 4x PV WMMA (d_v = 64).
// ---------------------------------------------------------------------------
__global__ void __launch_bounds__(128) attn_kernel(const _Float16* __restrict__ qA,
                                                   const _Float16* __restrict__ kB,
                                                   const _Float16* __restrict__ vB,
                                                   float* __restrict__ U) {
  __shared__ _Float16 pbuf[4][16 * 32];
  int lane = threadIdx.x & 31;
  int wave = threadIdx.x >> 5;
  int bh = blockIdx.x / 25;
  int nt = (blockIdx.x % 25) * 4 + wave;
  int b = bh >> 2, h = bh & 3;
  int r0 = lane & 15, half = lane >> 4;
  _Float16* pl = &pbuf[wave][0];

  v16h aq = *(const v16h*)(qA + (bh * 100 + nt) * 512 + lane * 16);
  float mrow[8], lrow[8];
#pragma unroll
  for (int r = 0; r < 8; ++r) { mrow[r] = -1e30f; lrow[r] = 0.f; }
  v8f acc[4] = {};

  for (int mc = 0; mc < 50; ++mc) {
    v8f s[2];
#pragma unroll
    for (int tp = 0; tp < 2; ++tp) {
      v16h bk = *(const v16h*)(kB + (bh * 100 + mc * 2 + tp) * 512 + lane * 16);
      v8f z = {};
      s[tp] = __builtin_amdgcn_wmma_f32_16x16x32_f16(false, aq, false, bk, (short)0, z, false, false);
    }
#pragma unroll
    for (int r = 0; r < 8; ++r) {
      float rm = fmaxf(s[0][r], s[1][r]);
      rm = fmaxf(rm, __shfl_xor(rm, 1, 32));
      rm = fmaxf(rm, __shfl_xor(rm, 2, 32));
      rm = fmaxf(rm, __shfl_xor(rm, 4, 32));
      rm = fmaxf(rm, __shfl_xor(rm, 8, 32));
      float nm = fmaxf(mrow[r], rm);
      float corr = __expf(mrow[r] - nm);
      mrow[r] = nm;
      float p0 = __expf(s[0][r] - nm);
      float p1 = __expf(s[1][r] - nm);
      float rs = p0 + p1;
      rs += __shfl_xor(rs, 1, 32);
      rs += __shfl_xor(rs, 2, 32);
      rs += __shfl_xor(rs, 4, 32);
      rs += __shfl_xor(rs, 8, 32);
      lrow[r] = lrow[r] * corr + rs;
#pragma unroll
      for (int dt = 0; dt < 4; ++dt) acc[dt][r] *= corr;
      int row = r + half * 8;
      pl[row * 32 + r0]      = (_Float16)p0;
      pl[row * 32 + 16 + r0] = (_Float16)p1;
    }
    asm volatile("s_wait_dscnt 0" ::: "memory");   // cross-lane LDS visibility
    int abase = r0 * 32 + half * 8;
    v8h a0 = *(const v8h*)(pl + abase);
    v8h a1 = *(const v8h*)(pl + abase + 16);
    v16h ap;
#pragma unroll
    for (int i = 0; i < 8; ++i) { ap[i] = a0[i]; ap[i + 8] = a1[i]; }
#pragma unroll
    for (int dt = 0; dt < 4; ++dt) {
      v16h bv = *(const v16h*)(vB + (((bh * 50 + mc) * 4) + dt) * 512 + lane * 16);
      acc[dt] = __builtin_amdgcn_wmma_f32_16x16x32_f16(false, ap, false, bv, (short)0, acc[dt], false, false);
    }
  }
#pragma unroll
  for (int dt = 0; dt < 4; ++dt) {
    int ch = h * 64 + dt * 16 + r0;
#pragma unroll
    for (int r = 0; r < 8; ++r) {
      U[(b * CHN + ch) * HW + nt * 16 + r + half * 8] = acc[dt][r] / lrow[r];
    }
  }
}

// ---------------------------------------------------------------------------
// 6) Depthwise 3x3 + BN + add attention out; emit proj GEMM B operand (f16).
// ---------------------------------------------------------------------------
__global__ void dwpack_kernel(const float* __restrict__ U, const float* __restrict__ qkv,
                              const float* __restrict__ wdwf, const float* __restrict__ bias_dw,
                              _Float16* __restrict__ UB) {
  int t = blockIdx.x * blockDim.x + threadIdx.x;
  int j = t & 15, lane = (t >> 4) & 31, rest = t >> 9;
  int nt = rest % 100; rest /= 100;
  int kt = rest % 8;  int b = rest / 8;
  int ci = kt * 32 + ((j < 8) ? 0 : 16) + ((lane >> 4) << 3) + (j & 7);
  int n  = nt * 16 + (lane & 15);
  int y = n / 40, x = n % 40;
  int h = ci >> 6, d = ci & 63;
  const float* vimg = qkv + (b * QKVC + h * 128 + 64 + d) * HW;
  float acc = bias_dw[ci];
#pragma unroll
  for (int ky = 0; ky < 3; ++ky) {
    int yy = y + ky - 1;
    if (yy < 0 || yy >= 40) continue;
#pragma unroll
    for (int kx = 0; kx < 3; ++kx) {
      int xx = x + kx - 1;
      if (xx < 0 || xx >= 40) continue;
      acc += wdwf[ci * 9 + ky * 3 + kx] * vimg[yy * 40 + xx];
    }
  }
  acc += U[(b * CHN + ci) * HW + n];
  UB[t] = (_Float16)acc;
}

// ---------------------------------------------------------------------------
// 7) Projection GEMM: (256x256) x (256x1600) per batch, final f32 output.
// ---------------------------------------------------------------------------
__global__ void __launch_bounds__(128) proj_gemm_kernel(const _Float16* __restrict__ wh,
                                                        const _Float16* __restrict__ UB,
                                                        const float* __restrict__ bias,
                                                        float* __restrict__ out) {
  int lane = threadIdx.x & 31;
  int wid  = blockIdx.x * 4 + (threadIdx.x >> 5);
  int nt   = wid % 100;
  int cot  = (wid / 100) % 16;
  int b    = wid / 1600;
  int r0 = lane & 15, half = lane >> 4;
  v8f c = {};
#pragma unroll
  for (int kt = 0; kt < 8; ++kt) {
    const _Float16* ap = wh + (cot * 16 + r0) * CHN + kt * 32 + half * 8;
    v8h a0 = *(const v8h*)(ap);
    v8h a1 = *(const v8h*)(ap + 16);
    v16h a;
#pragma unroll
    for (int i = 0; i < 8; ++i) { a[i] = a0[i]; a[i + 8] = a1[i]; }
    v16h bm = *(const v16h*)(UB + ((b * 8 + kt) * 100 + nt) * 512 + lane * 16);
    c = __builtin_amdgcn_wmma_f32_16x16x32_f16(false, a, false, bm, (short)0, c, false, false);
  }
#pragma unroll
  for (int r = 0; r < 8; ++r) {
    int co = cot * 16 + r + half * 8;
    out[(b * CHN + co) * HW + nt * 16 + r0] = c[r] + bias[co];
  }
}

// ---------------------------------------------------------------------------
extern "C" void kernel_launch(void* const* d_in, const int* in_sizes, int n_in,
                              void* d_out, int out_size, void* d_ws, size_t ws_size,
                              hipStream_t stream) {
  (void)in_sizes; (void)n_in; (void)out_size; (void)ws_size;
  const float* x      = (const float*)d_in[0];
  const float* w_qkv  = (const float*)d_in[1];
  const float* g_qkv  = (const float*)d_in[2];
  const float* b_qkv  = (const float*)d_in[3];
  const float* m_qkv  = (const float*)d_in[4];
  const float* v_qkv  = (const float*)d_in[5];
  const float* w_dw   = (const float*)d_in[6];
  const float* g_dw   = (const float*)d_in[7];
  const float* b_dw   = (const float*)d_in[8];
  const float* m_dw   = (const float*)d_in[9];
  const float* v_dw   = (const float*)d_in[10];
  const float* w_proj = (const float*)d_in[11];
  const float* g_proj = (const float*)d_in[12];
  const float* b_proj = (const float*)d_in[13];
  const float* m_proj = (const float*)d_in[14];
  const float* v_proj = (const float*)d_in[15];
  float* out = (float*)d_out;

  char* ws = (char*)d_ws;
  auto carve = [&](size_t bytes) -> char* {
    char* p = ws;
    ws += (bytes + 255) & ~(size_t)255;
    return p;
  };
  float*    qkv       = (float*)   carve((size_t)NB * QKVC * HW * 4);   // 13.1 MB
  float*    U         = (float*)   carve((size_t)NB * CHN * HW * 4);    //  6.6 MB
  _Float16* wqkv_h    = (_Float16*)carve((size_t)QKVC * CHN * 2);
  _Float16* wproj_h   = (_Float16*)carve((size_t)CHN * CHN * 2);
  float*    wdwf      = (float*)   carve((size_t)CHN * 9 * 4);
  float*    bias_qkv  = (float*)   carve((size_t)QKVC * 4);
  float*    bias_proj = (float*)   carve((size_t)CHN * 4);
  float*    bias_dw   = (float*)   carve((size_t)CHN * 4);
  _Float16* xB        = (_Float16*)carve((size_t)NB * 8 * 100 * 512 * 2);      // 3.3 MB
  _Float16* qA        = (_Float16*)carve((size_t)16 * 100 * 512 * 2);          // 1.6 MB
  _Float16* kBp       = (_Float16*)carve((size_t)16 * 100 * 512 * 2);          // 1.6 MB
  _Float16* vBp       = (_Float16*)carve((size_t)16 * 50 * 4 * 512 * 2);       // 3.3 MB
  _Float16* UB        = (_Float16*)carve((size_t)NB * 8 * 100 * 512 * 2);      // 3.3 MB

  // 1) BN folding + weight conversion
  {
    int total = QKVC * CHN + CHN * CHN + CHN * 9 + QKVC + CHN + CHN;
    prep_kernel<<<(total + 255) / 256, 256, 0, stream>>>(
        w_qkv, g_qkv, b_qkv, m_qkv, v_qkv,
        w_dw, g_dw, b_dw, m_dw, v_dw,
        w_proj, g_proj, b_proj, m_proj, v_proj,
        wqkv_h, bias_qkv, wproj_h, bias_proj, wdwf, bias_dw);
  }
  // 2) pack x into B-operand layout
  pack_x_kernel<<<(NB * 8 * 100 * 512) / 256, 256, 0, stream>>>(x, xB);
  // 3) QKV GEMM (12800 waves)
  qkv_gemm_kernel<<<3200, 128, 0, stream>>>(wqkv_h, xB, bias_qkv, qkv);
  // 4) repack q/k/v into WMMA operand layouts
  repack_qkv_kernel<<<(16 * 100 * 512 * 2 + 16 * 50 * 4 * 512) / 256, 256, 0, stream>>>(
      qkv, qA, kBp, vBp);
  // 5) flash attention (1600 waves x 50 key-chunks)
  attn_kernel<<<400, 128, 0, stream>>>(qA, kBp, vBp, U);
  // 6) depthwise conv + add + pack proj B operand
  dwpack_kernel<<<(NB * 8 * 100 * 512) / 256, 256, 0, stream>>>(U, qkv, wdwf, bias_dw, UB);
  // 7) projection GEMM -> final output
  proj_gemm_kernel<<<1600, 128, 0, stream>>>(wproj_h, UB, bias_proj, out);
}
